// MultiHeadAttention_25194278158766
// MI455X (gfx1250) — matrix operand order repaired and checked by reading, not compiled, not run on hardware
//
#include <hip/hip_runtime.h>
#include <hip/hip_bf16.h>

// MHA forward for MI455X (gfx1250, wave32, WMMA + async global->LDS).
// Pipeline: [QKV proj f16-WMMA GEMM x3] -> [flash attention f16-WMMA] -> [out proj GEMM].
// Workspace layout (requires ws_size >= 40 MB):
//   [0,8MB)   q heads f16  [B*H, S, 64]
//   [8,16MB)  k heads f16
//   [16,24MB) v heads f16
//   [24,40MB) attention context f32 [B*S, H*DV]

typedef _Float16 f16_t;
typedef __attribute__((ext_vector_type(16))) _Float16 v16h;
typedef __attribute__((ext_vector_type(8)))  float    v8f;

union FragA { v16h v; unsigned int u[8]; };

static constexpr int BB  = 2;
static constexpr int SS  = 2048;
static constexpr int DD  = 1024;
static constexpr int HH  = 16;
static constexpr int DKV = 64;

// ---------------------------------------------------------------------------
// GEMM: C[4096,1024] = A[4096,1024](f32) @ W[1024,1024](f32) + bias
// MODE 0: store f16, head-split into [B*H, S, 64]
// MODE 1: store f32 row-major [4096,1024]
// Workgroup tile 128x128, 8 waves (4M x 2N), wave tile 32x64 (2x4 WMMA accs),
// K staged in 32-deep LDS chunks with on-the-fly f32->f16 conversion.
// ---------------------------------------------------------------------------
template <int MODE>
__global__ __launch_bounds__(256)
void gemm_bias_kernel(const float* __restrict__ A,
                      const float* __restrict__ W,
                      const float* __restrict__ bias,
                      f16_t* __restrict__ outH,
                      float* __restrict__ outF)
{
    __shared__ f16_t Ash[128][34];   // [m][k], +2 pad
    __shared__ f16_t Wsh[128][34];   // [n][k] (transposed), +2 pad

    const int tid   = threadIdx.x;
    const int lane  = tid & 31;
    const int wave  = tid >> 5;
    const int lh    = lane & 15;
    const int hi    = lane >> 4;           // lane half: 0 or 1
    const int waveM = wave & 3;            // 0..3
    const int waveN = wave >> 2;           // 0..1
    const int mBase = blockIdx.x * 128;
    const int nBase = blockIdx.y * 128;
    const int rowBase = waveM * 32;
    const int colBase = waveN * 64;

    v8f acc[2][4] = {};

    for (int k0 = 0; k0 < DD; k0 += 32) {
        __syncthreads();
        // Stage A chunk 128x32: 1024 float4, 4 per thread, cvt f32->f16.
        #pragma unroll
        for (int i = 0; i < 4; ++i) {
            const int idx = tid * 4 + i;            // 0..1023
            const int row = idx >> 3;
            const int c4  = (idx & 7) << 2;
            const float* src = A + (size_t)(mBase + row) * DD + k0 + c4;
            if (k0 + 32 < DD) __builtin_prefetch(src + 32, 0, 0);  // global_prefetch_b8
            const float4 f = *(const float4*)src;
            Ash[row][c4 + 0] = (f16_t)f.x;
            Ash[row][c4 + 1] = (f16_t)f.y;
            Ash[row][c4 + 2] = (f16_t)f.z;
            Ash[row][c4 + 3] = (f16_t)f.w;
        }
        // Stage W chunk 32x128 transposed: 1024 float4, 4 per thread.
        #pragma unroll
        for (int i = 0; i < 4; ++i) {
            const int idx = tid * 4 + i;            // 0..1023
            const int k   = idx >> 5;               // 0..31
            const int n4  = (idx & 31) << 2;        // 0..124
            const float* src = W + (size_t)(k0 + k) * DD + nBase + n4;
            if (k0 + 32 < DD) __builtin_prefetch(src + 32 * DD, 0, 0);
            const float4 f = *(const float4*)src;
            Wsh[n4 + 0][k] = (f16_t)f.x;
            Wsh[n4 + 1][k] = (f16_t)f.y;
            Wsh[n4 + 2][k] = (f16_t)f.z;
            Wsh[n4 + 3][k] = (f16_t)f.w;
        }
        __syncthreads();

        // A fragment per ISA: lanes 0-15 K=0..7,16..23 ; lanes 16-31 K=8..15,24..31
        FragA af[2], bf[4];
        #pragma unroll
        for (int r = 0; r < 2; ++r) {
            const unsigned int* ap = (const unsigned int*)&Ash[rowBase + r * 16 + lh][0];
            #pragma unroll
            for (int e = 0; e < 4; ++e) {
                af[r].u[e]     = ap[hi * 4 + e];
                af[r].u[4 + e] = ap[hi * 4 + 8 + e];
            }
        }
        // B fragment (same K order as A): lanes 0-15 K=0..7,16..23 ; lanes 16-31 K=8..15,24..31 (N = lane&15)
        #pragma unroll
        for (int c = 0; c < 4; ++c) {
            const unsigned int* wp = (const unsigned int*)&Wsh[colBase + c * 16 + lh][0];
            #pragma unroll
            for (int e = 0; e < 4; ++e) {
                bf[c].u[e]     = wp[hi * 4 + e];
                bf[c].u[4 + e] = wp[hi * 4 + 8 + e];
            }
        }
        #pragma unroll
        for (int r = 0; r < 2; ++r)
            #pragma unroll
            for (int c = 0; c < 4; ++c)
                acc[r][c] = __builtin_amdgcn_wmma_f32_16x16x32_f16(
                    false, af[r].v, false, bf[c].v, (short)0, acc[r][c], false, false);
    }

    // Epilogue. C layout: VGPR i -> M = i + 8*hi ; N = lane&15.
    #pragma unroll
    for (int r = 0; r < 2; ++r) {
        #pragma unroll
        for (int c = 0; c < 4; ++c) {
            const int colG = nBase + colBase + c * 16 + lh;
            const float bv = bias[colG];
            #pragma unroll
            for (int i = 0; i < 8; ++i) {
                const int rowG = mBase + rowBase + r * 16 + i + hi * 8;
                const float v = acc[r][c][i] + bv;
                if (MODE == 0) {
                    const int h = colG >> 6, d = colG & 63;
                    const int b = rowG >> 11, s = rowG & (SS - 1);
                    outH[((size_t)(b * HH + h) * SS + s) * DKV + d] = (f16_t)v;
                } else {
                    outF[(size_t)rowG * DD + colG] = v;
                }
            }
        }
    }
}

// ---------------------------------------------------------------------------
// Flash attention: one workgroup per (b*H+h, 128-row Q block); each wave owns
// 16 Q rows. K tiles staged global->LDS with ASYNC loads (no VGPR round-trip,
// tracked on ASYNCcnt); V transposed through VGPRs. Online softmax in fp32.
// ---------------------------------------------------------------------------
__global__ __launch_bounds__(256)
void flash_attn_kernel(const f16_t* __restrict__ Qh,
                       const f16_t* __restrict__ Kh,
                       const f16_t* __restrict__ Vh,
                       const int*   __restrict__ padMask,
                       float* __restrict__ O)       // [B*S, H*DV] f32
{
    __shared__ f16_t Ksh[64][66];       // [key][d]
    __shared__ f16_t Vsh[64][66];       // [d][key] (transposed)
    __shared__ f16_t Psh[8][16][66];    // per-wave P relayout scratch
    __shared__ int   Msh[64];

    const int tid  = threadIdx.x;
    const int lane = tid & 31;
    const int wave = tid >> 5;
    const int lh   = lane & 15;
    const int hi   = lane >> 4;
    const int qb   = blockIdx.x;                 // 0..15
    const int bh   = blockIdx.y;                 // 0..31
    const int b    = bh >> 4;
    const int h    = bh & (HH - 1);
    const int qBase = qb * 128 + wave * 16;      // this wave's 16 query rows

    // Uniform 64-bit base of this head's K rows (SGPR pair for async GVS mode).
    const unsigned long long kRowBase =
        (unsigned long long)(uintptr_t)(Kh + (size_t)bh * SS * DKV);

    // Resident Q fragments: q is A-matrix 16x64 -> two K=32 chunks.
    FragA qf[2];
    {
        const unsigned int* qp =
            (const unsigned int*)(Qh + ((size_t)bh * SS + qBase + lh) * DKV);
        #pragma unroll
        for (int kc = 0; kc < 2; ++kc) {
            #pragma unroll
            for (int e = 0; e < 4; ++e) {
                qf[kc].u[e]     = qp[kc * 16 + hi * 4 + e];
                qf[kc].u[4 + e] = qp[kc * 16 + hi * 4 + 8 + e];
            }
        }
    }

    v8f oacc[4] = {};
    float m_i[8], l_i[8];
    #pragma unroll
    for (int i = 0; i < 8; ++i) { m_i[i] = -1e30f; l_i[i] = 0.f; }

    const int nkt = 2 * (qb + 1);   // causal: skip key tiles entirely above block
    for (int kt = 0; kt < nkt; ++kt) {
        const int keyBase = kt * 64;
        __syncthreads();
        // K tile 64x64 f16: async global->LDS, 16B per lane, 2 issues per thread.
        // GVS addressing: saddr = head base (SGPR pair), vaddr = 32-bit byte offset,
        // vdst = LDS byte address (low 32 bits of generic shared address).
        #pragma unroll
        for (int i = 0; i < 2; ++i) {
            const int idx = tid * 2 + i;
            const int key = idx >> 3;
            const int d   = (idx & 7) << 3;       // 8 halves = 16 bytes
            const unsigned lds  =
                (unsigned)(unsigned long long)(uintptr_t)&Ksh[key][d];
            const unsigned goff =
                (unsigned)(((unsigned)(keyBase + key) * DKV + d) * sizeof(f16_t));
            asm volatile("global_load_async_to_lds_b128 %0, %1, %2 offset:0"
                         :: "v"(lds), "v"(goff), "s"(kRowBase) : "memory");
        }
        // V tile transposed into [d][key] (needs VGPR pass for the transpose).
        #pragma unroll
        for (int i = 0; i < 8; ++i) {
            const int idx = tid * 8 + i;            // 0..2047
            const int key = idx >> 5;
            const int d   = (idx & 31) << 1;
            union { unsigned int u; f16_t hh[2]; } t;
            t.u = *(const unsigned int*)(Vh + ((size_t)bh * SS + keyBase + key) * DKV + d);
            Vsh[d][key]     = t.hh[0];
            Vsh[d + 1][key] = t.hh[1];
        }
        if (tid < 64) Msh[tid] = padMask[b * SS + keyBase + tid];
        asm volatile("s_wait_asynccnt 0x0" ::: "memory");   // K tile landed in LDS
        __syncthreads();

        // Scores: 4 key sub-tiles of 16, each 2 WMMA over DK=64.
        v8f sv[4];
        #pragma unroll
        for (int j = 0; j < 4; ++j) {
            v8f s = {};
            const unsigned int* kp = (const unsigned int*)&Ksh[j * 16 + lh][0];
            #pragma unroll
            for (int kc = 0; kc < 2; ++kc) {
                FragA bf;
                #pragma unroll
                for (int e = 0; e < 4; ++e) {
                    bf.u[e]     = kp[kc * 16 + hi * 4 + e];
                    bf.u[4 + e] = kp[kc * 16 + hi * 4 + 8 + e];
                }
                s = __builtin_amdgcn_wmma_f32_16x16x32_f16(
                    false, qf[kc].v, false, bf.v, (short)0, s, false, false);
            }
            sv[j] = s;
        }

        // Scale + pad/causal mask (matches reference: s - 1e9*mask).
        #pragma unroll
        for (int j = 0; j < 4; ++j) {
            const int key = keyBase + j * 16 + lh;
            const int pm  = Msh[j * 16 + lh];
            #pragma unroll
            for (int i = 0; i < 8; ++i) {
                const int qrow = qBase + i + hi * 8;
                const float mk = ((key > qrow) || (pm != 0)) ? 1.f : 0.f;
                sv[j][i] = sv[j][i] * 0.125f - 1e9f * mk;
            }
        }

        // Online softmax: each score row lives in 16 lanes of one lane-half.
        #pragma unroll
        for (int i = 0; i < 8; ++i) {
            float rm = fmaxf(fmaxf(sv[0][i], sv[1][i]), fmaxf(sv[2][i], sv[3][i]));
            #pragma unroll
            for (int off = 1; off < 16; off <<= 1)
                rm = fmaxf(rm, __shfl_xor(rm, off, 16));
            const float mnew  = fmaxf(m_i[i], rm);
            const float alpha = __expf(m_i[i] - mnew);
            m_i[i] = mnew;
            float rs = 0.f;
            #pragma unroll
            for (int j = 0; j < 4; ++j) {
                const float p = __expf(sv[j][i] - mnew);
                sv[j][i] = p;
                rs += p;
            }
            #pragma unroll
            for (int off = 1; off < 16; off <<= 1)
                rs += __shfl_xor(rs, off, 16);
            l_i[i] = l_i[i] * alpha + rs;
            #pragma unroll
            for (int jd = 0; jd < 4; ++jd) oacc[jd][i] *= alpha;
        }

        // Relayout P (C-layout f32) -> A-fragment f16 via wave-private LDS.
        #pragma unroll
        for (int j = 0; j < 4; ++j)
            #pragma unroll
            for (int i = 0; i < 8; ++i)
                Psh[wave][i + hi * 8][j * 16 + lh] = (f16_t)sv[j][i];
        __builtin_amdgcn_wave_barrier();  // DS ops in-order per wave; block reordering

        // O += P @ V : contraction over 64 keys (two K=32 chunks), 4 d-tiles.
        #pragma unroll
        for (int kc = 0; kc < 2; ++kc) {
            FragA pf;
            const unsigned int* pp = (const unsigned int*)&Psh[wave][lh][0];
            #pragma unroll
            for (int e = 0; e < 4; ++e) {
                pf.u[e]     = pp[kc * 16 + hi * 4 + e];
                pf.u[4 + e] = pp[kc * 16 + hi * 4 + 8 + e];
            }
            #pragma unroll
            for (int jd = 0; jd < 4; ++jd) {
                FragA vf;
                const unsigned int* vp = (const unsigned int*)&Vsh[jd * 16 + lh][0];
                #pragma unroll
                for (int e = 0; e < 4; ++e) {
                    vf.u[e]     = vp[kc * 16 + hi * 4 + e];
                    vf.u[4 + e] = vp[kc * 16 + hi * 4 + 8 + e];
                }
                oacc[jd] = __builtin_amdgcn_wmma_f32_16x16x32_f16(
                    false, pf.v, false, vf.v, (short)0, oacc[jd], false, false);
            }
        }
    }

    // Normalize and store context into [B*S, H*DV] f32 for the output GEMM.
    #pragma unroll
    for (int i = 0; i < 8; ++i) {
        const int   qrow = qBase + i + hi * 8;
        const float inv  = 1.f / l_i[i];
        #pragma unroll
        for (int jd = 0; jd < 4; ++jd)
            O[(size_t)(b * SS + qrow) * (HH * DKV) + h * DKV + jd * 16 + lh] =
                oacc[jd][i] * inv;
    }
}

// ---------------------------------------------------------------------------
extern "C" void kernel_launch(void* const* d_in, const int* in_sizes, int n_in,
                              void* d_out, int out_size, void* d_ws, size_t ws_size,
                              hipStream_t stream) {
    const float* Q  = (const float*)d_in[0];
    const float* K  = (const float*)d_in[1];
    const float* V  = (const float*)d_in[2];
    const int*   pm = (const int*)  d_in[3];
    const float* Wq = (const float*)d_in[4];
    const float* bq = (const float*)d_in[5];
    const float* Wk = (const float*)d_in[6];
    const float* bk = (const float*)d_in[7];
    const float* Wv = (const float*)d_in[8];
    const float* bv = (const float*)d_in[9];
    const float* Wo = (const float*)d_in[10];
    const float* bo = (const float*)d_in[11];
    float* out = (float*)d_out;

    char* ws = (char*)d_ws;
    f16_t* qh = (f16_t*)(ws + ((size_t)0  << 20));
    f16_t* kh = (f16_t*)(ws + ((size_t)8  << 20));
    f16_t* vh = (f16_t*)(ws + ((size_t)16 << 20));
    float* Ob = (float*)(ws + ((size_t)24 << 20));

    const dim3 blk(256);
    const dim3 gG(32, 8, 1);    // 4096/128 x 1024/128
    const dim3 gA(16, 32, 1);   // S/128 x B*H

    gemm_bias_kernel<0><<<gG, blk, 0, stream>>>(Q, Wq, bq, qh, nullptr);
    gemm_bias_kernel<0><<<gG, blk, 0, stream>>>(K, Wk, bk, kh, nullptr);
    gemm_bias_kernel<0><<<gG, blk, 0, stream>>>(V, Wv, bv, vh, nullptr);
    flash_attn_kernel<<<gA, blk, 0, stream>>>(qh, kh, vh, pm, Ob);
    gemm_bias_kernel<1><<<gG, blk, 0, stream>>>(Ob, Wo, bo, nullptr, out);
}
